// GINDecoder_84284438217359
// MI455X (gfx1250) — matrix operand order repaired and checked by reading, not compiled, not run on hardware
//
#include <hip/hip_runtime.h>
#include <hip/hip_bf16.h>

// ---------- vector types for WMMA ----------
typedef __attribute__((ext_vector_type(16))) __bf16 v16bf;
typedef __attribute__((ext_vector_type(8)))  __bf16 v8bf;   // 16 bytes
typedef __attribute__((ext_vector_type(8)))  float  v8f;

// ---------- constants ----------
#define NODES     32000
#define EDGES     512000
#define D_IN      256
#define D_H       512
#define D_E       256
#define NGRAPH    16
#define NODE_NUM  2000
#define BN_EPS    1e-5f

// fragment = 16x32 (A) or 32x16 (B) bf16 tile slice per lane
union Frag { v16bf v; v8bf h[2]; };

// A (16x32 bf16): lane = kh*16+m; per-lane contiguous K range [kh*16, kh*16+16)
__device__ __forceinline__ void load_a(Frag& f, const __bf16* p) {
    f.h[0] = *(const v8bf*)(p);
    f.h[1] = *(const v8bf*)(p + 8);
}
// B (32x16 bf16): lane = kh*16+n; per-lane contiguous K range [kh*16, kh*16+16)
__device__ __forceinline__ void load_b(Frag& f, const __bf16* p) {
    f.h[0] = *(const v8bf*)(p);
    f.h[1] = *(const v8bf*)(p + 8);
}

// ======================================================================
// fp32 -> bf16 conversion (grid-stride)
// ======================================================================
__global__ void k_f32_to_bf16(const float* __restrict__ in, __bf16* __restrict__ out, long long n) {
    long long i = (long long)blockIdx.x * blockDim.x + threadIdx.x;
    long long stride = (long long)gridDim.x * blockDim.x;
    for (; i < n; i += stride) out[i] = (__bf16)in[i];
}

// ======================================================================
// zero a float buffer
// ======================================================================
__global__ void k_zero(float* __restrict__ p, long long n) {
    long long i = (long long)blockIdx.x * blockDim.x + threadIdx.x;
    long long stride = (long long)gridDim.x * blockDim.x;
    for (; i < n; i += stride) p[i] = 0.0f;
}

// ======================================================================
// WMMA bf16 GEMM:  C[M,N] = A[M,K] @ W[N,K]^T + bias[N]
//   one wave per block; wave computes a 32(M) x 64(N) strip
//   (2 M-subtiles x 4 N-subtiles = 8 f32 accumulators)
//   K-loop unrolled 2x with double-buffered fragments so loads for step
//   k+32 are in flight while WMMAs consume step k.
//   grid = (M/32, N/64), block = 32
// ======================================================================
__global__ __launch_bounds__(32)
void k_gemm_bf16(const __bf16* __restrict__ A, const __bf16* __restrict__ W,
                 const float* __restrict__ bias, float* __restrict__ C,
                 int M, int N, int K) {
    const int lane = threadIdx.x & 31;
    const int m0   = blockIdx.x * 32;
    const int n0   = blockIdx.y * 64;
    const int row  = lane & 15;       // M (for A) / N (for B) within 16-tile
    const int kh   = lane >> 4;       // K-half selector

    v8f acc[2][4] = {};

    const __bf16* arow[2];
#pragma unroll
    for (int mt = 0; mt < 2; ++mt)
        arow[mt] = A + (size_t)(m0 + mt * 16 + row) * K + kh * 16;
    const __bf16* brow[4];
#pragma unroll
    for (int nt = 0; nt < 4; ++nt)
        brow[nt] = W + (size_t)(n0 + nt * 16 + row) * K + kh * 16;

    Frag a0[2], a1[2], b0[4], b1[4];

    // preload K-step 0 into buffer 0
#pragma unroll
    for (int mt = 0; mt < 2; ++mt) load_a(a0[mt], arow[mt]);
#pragma unroll
    for (int nt = 0; nt < 4; ++nt) load_b(b0[nt], brow[nt]);

    for (int k0 = 0; k0 < K; k0 += 64) {   // K is a multiple of 64 (256 / 512)
        // prefetch far-ahead A stream (speculative; dropped if OOB)
        __builtin_prefetch(arow[0] + k0 + 256, 0, 0);
        __builtin_prefetch(arow[1] + k0 + 256, 0, 0);

        // issue loads for k0+32 into buffer 1
#pragma unroll
        for (int mt = 0; mt < 2; ++mt) load_a(a1[mt], arow[mt] + k0 + 32);
#pragma unroll
        for (int nt = 0; nt < 4; ++nt) load_b(b1[nt], brow[nt] + k0 + 32);

        // compute with buffer 0 (its loads were issued last iteration)
#pragma unroll
        for (int mt = 0; mt < 2; ++mt)
#pragma unroll
            for (int nt = 0; nt < 4; ++nt)
                acc[mt][nt] = __builtin_amdgcn_wmma_f32_16x16x32_bf16(
                    false, a0[mt].v, false, b0[nt].v, (short)0, acc[mt][nt], false, false);

        // issue loads for k0+64 into buffer 0 (skip past end)
        if (k0 + 64 < K) {
#pragma unroll
            for (int mt = 0; mt < 2; ++mt) load_a(a0[mt], arow[mt] + k0 + 64);
#pragma unroll
            for (int nt = 0; nt < 4; ++nt) load_b(b0[nt], brow[nt] + k0 + 64);
        }

        // compute with buffer 1
#pragma unroll
        for (int mt = 0; mt < 2; ++mt)
#pragma unroll
            for (int nt = 0; nt < 4; ++nt)
                acc[mt][nt] = __builtin_amdgcn_wmma_f32_16x16x32_bf16(
                    false, a1[mt].v, false, b1[nt].v, (short)0, acc[mt][nt], false, false);
    }

    // C/D layout: VGPR r -> M = r + 8*kh, N = lane&15
#pragma unroll
    for (int mt = 0; mt < 2; ++mt) {
#pragma unroll
        for (int nt = 0; nt < 4; ++nt) {
            int n = n0 + nt * 16 + (lane & 15);
            float bv = bias[n];
#pragma unroll
            for (int r = 0; r < 8; ++r) {
                int m = m0 + mt * 16 + r + 8 * kh;
                C[(size_t)m * N + n] = acc[mt][nt][r] + bv;
            }
        }
    }
}

// ======================================================================
// edge scatter-add: Agg[dst[e], :] += H[src[e], :]   (float4 per thread)
// ======================================================================
__global__ void k_scatter(const float* __restrict__ H, const int* __restrict__ src,
                          const int* __restrict__ dst, float* __restrict__ Agg,
                          int E, int D) {
    long long idx = (long long)blockIdx.x * blockDim.x + threadIdx.x;
    int chunks = D >> 2;
    long long total = (long long)E * chunks;
    if (idx >= total) return;
    int e = (int)(idx / chunks);
    int c = (int)(idx % chunks) * 4;
    int s = src[e], d = dst[e];
    const float4 v = *(const float4*)(H + (size_t)s * D + c);
    float* p = Agg + (size_t)d * D + c;
    atomicAdd(p + 0, v.x);
    atomicAdd(p + 1, v.y);
    atomicAdd(p + 2, v.z);
    atomicAdd(p + 3, v.w);
}

// ======================================================================
// H = relu(H + Agg)   (GIN: agg + (1+eps)*h, eps = 0)
// ======================================================================
__global__ void k_combine(float* __restrict__ H, const float* __restrict__ Agg, long long n) {
    long long i = (long long)blockIdx.x * blockDim.x + threadIdx.x;
    long long stride = (long long)gridDim.x * blockDim.x;
    for (; i < n; i += stride) {
        float v = H[i] + Agg[i];
        H[i] = v > 0.0f ? v : 0.0f;
    }
}

// ======================================================================
// per-column mean / rstd  (one block per column; one-pass sum/sumsq, no atomics)
// ======================================================================
__global__ void k_colstats(const float* __restrict__ H, float* __restrict__ mu,
                           float* __restrict__ rstd, int rows, int D) {
    __shared__ float ssum[256];
    __shared__ float ssq[256];
    int col = blockIdx.x;
    float s = 0.0f, sq = 0.0f;
    for (int r = threadIdx.x; r < rows; r += blockDim.x) {
        float v = H[(size_t)r * D + col];
        s += v; sq += v * v;
    }
    ssum[threadIdx.x] = s; ssq[threadIdx.x] = sq;
    __syncthreads();
    for (int off = 128; off > 0; off >>= 1) {
        if ((int)threadIdx.x < off) {
            ssum[threadIdx.x] += ssum[threadIdx.x + off];
            ssq[threadIdx.x]  += ssq[threadIdx.x + off];
        }
        __syncthreads();
    }
    if (threadIdx.x == 0) {
        float m = ssum[0] / (float)rows;
        float v = ssq[0] / (float)rows - m * m;
        mu[col]   = m;
        rstd[col] = rsqrtf(v + BN_EPS);
    }
}

// ======================================================================
// batchnorm + leakyrelu(0.1) -> bf16 activation (+ optional f32 copy)
// ======================================================================
__global__ void k_bn_act(const float* __restrict__ H, const float* __restrict__ mu,
                         const float* __restrict__ rstd, const float* __restrict__ g,
                         const float* __restrict__ b, __bf16* __restrict__ actb,
                         float* __restrict__ actf, long long n, int D) {
    long long i = (long long)blockIdx.x * blockDim.x + threadIdx.x;
    long long stride = (long long)gridDim.x * blockDim.x;
    for (; i < n; i += stride) {
        int c = (int)(i % D);
        float y = g[c] * (H[i] - mu[c]) * rstd[c] + b[c];
        y = y >= 0.0f ? y : 0.1f * y;
        actb[i] = (__bf16)y;
        if (actf) actf[i] = y;
    }
}

// ======================================================================
// per-graph mean pool over 2000 nodes + 256->2 head + argmax
//   grid = 16 (graphs), block = 256 (one thread per feature)
//   out layout: [16*2] logits then [16] argmax (as float)
// ======================================================================
__global__ void k_pool_head(const float* __restrict__ act, const float* __restrict__ Wp,
                            const float* __restrict__ bp, float* __restrict__ out) {
    __shared__ float gvec[256];
    __shared__ float red[256];
    int gi = blockIdx.x;
    int t  = threadIdx.x;
    const float* base = act + (size_t)gi * NODE_NUM * D_E;
    float s = 0.0f;
    for (int r = 0; r < NODE_NUM; ++r) s += base[(size_t)r * D_E + t];
    gvec[t] = s / (float)NODE_NUM;
    __syncthreads();
    float o0 = 0.0f, o1 = 0.0f;
    for (int c = 0; c < 2; ++c) {
        red[t] = gvec[t] * Wp[c * D_E + t];
        __syncthreads();
        for (int off = 128; off > 0; off >>= 1) {
            if (t < off) red[t] += red[t + off];
            __syncthreads();
        }
        if (t == 0) { if (c == 0) o0 = red[0] + bp[0]; else o1 = red[0] + bp[1]; }
        __syncthreads();
    }
    if (t == 0) {
        out[gi * 2 + 0] = o0;
        out[gi * 2 + 1] = o1;
        out[NGRAPH * 2 + gi] = (o1 > o0) ? 1.0f : 0.0f;  // argmax, first-max wins ties
    }
}

// ======================================================================
// host driver
// ======================================================================
static inline long long cdivll(long long a, long long b) { return (a + b - 1) / b; }

extern "C" void kernel_launch(void* const* d_in, const int* in_sizes, int n_in,
                              void* d_out, int out_size, void* d_ws, size_t ws_size,
                              hipStream_t stream) {
    (void)in_sizes; (void)n_in; (void)out_size; (void)ws_size;

    const float* x   = (const float*)d_in[0];
    const int*   ei  = (const int*)d_in[1];          // [2, E]
    const float* W1  = (const float*)d_in[2];
    const float* b1  = (const float*)d_in[3];
    const float* W2  = (const float*)d_in[4];
    const float* b2  = (const float*)d_in[5];
    const float* W3  = (const float*)d_in[6];
    const float* b3  = (const float*)d_in[7];
    const float* g1  = (const float*)d_in[8];
    const float* be1 = (const float*)d_in[9];
    const float* g2  = (const float*)d_in[10];
    const float* be2 = (const float*)d_in[11];
    const float* g3  = (const float*)d_in[12];
    const float* be3 = (const float*)d_in[13];
    const float* Wp  = (const float*)d_in[14];
    const float* bp  = (const float*)d_in[15];
    float* out = (float*)d_out;

    const int* src = ei;            // edge_index[0]
    const int* dst = ei + EDGES;    // edge_index[1]

    // ---- workspace carve (bytes) ----
    char* ws = (char*)d_ws;
    const size_t OFF_ACTB = 0;                           // bf16 activations: 32000*512*2
    const size_t OFF_WB   = OFF_ACTB + (size_t)NODES * D_H * 2;
    const size_t OFF_HF   = OFF_WB   + (size_t)D_H * D_H * 2;          // fp32 GEMM out
    const size_t OFF_AGG  = OFF_HF   + (size_t)NODES * D_H * 4;        // fp32 scatter target
    const size_t OFF_MU   = OFF_AGG  + (size_t)NODES * D_H * 4;
    const size_t OFF_RSTD = OFF_MU   + 2048;
    __bf16* actb = (__bf16*)(ws + OFF_ACTB);
    __bf16* Wb   = (__bf16*)(ws + OFF_WB);
    float*  Hf   = (float*)(ws + OFF_HF);
    float*  Agg  = (float*)(ws + OFF_AGG);
    float*  mu   = (float*)(ws + OFF_MU);
    float*  rstd = (float*)(ws + OFF_RSTD);

    const int TB = 256;

    // input activations -> bf16
    {
        long long n = (long long)NODES * D_IN;
        k_f32_to_bf16<<<dim3((unsigned)cdivll(n, TB)), TB, 0, stream>>>(x, actb, n);
    }

    // ---- per-layer driver ----
    struct Layer { const float *W, *b, *g, *be; int din, dout; };
    Layer layers[3] = {
        { W1, b1, g1, be1, D_IN, D_H },
        { W2, b2, g2, be2, D_H,  D_H },
        { W3, b3, g3, be3, D_H,  D_E },
    };

    for (int L = 0; L < 3; ++L) {
        const Layer& ly = layers[L];
        const long long nW = (long long)ly.dout * ly.din;
        const long long nH = (long long)NODES * ly.dout;

        // weights -> bf16
        k_f32_to_bf16<<<dim3((unsigned)cdivll(nW, TB)), TB, 0, stream>>>(ly.W, Wb, nW);

        // H = act @ W^T + b   (WMMA bf16, f32 accumulate)
        dim3 ggrid(NODES / 32, ly.dout / 64);
        k_gemm_bf16<<<ggrid, 32, 0, stream>>>(actb, Wb, ly.b, Hf, NODES, ly.dout, ly.din);

        // Agg = 0; scatter-add over edges
        k_zero<<<dim3((unsigned)cdivll(nH, TB)), TB, 0, stream>>>(Agg, nH);
        long long st = (long long)EDGES * (ly.dout >> 2);
        k_scatter<<<dim3((unsigned)cdivll(st, TB)), TB, 0, stream>>>(Hf, src, dst, Agg, EDGES, ly.dout);

        // H = relu(H + Agg)
        k_combine<<<dim3((unsigned)cdivll(nH, TB)), TB, 0, stream>>>(Hf, Agg, nH);

        // batchnorm stats (block per column)
        k_colstats<<<dim3(ly.dout), 256, 0, stream>>>(Hf, mu, rstd, NODES, ly.dout);

        // bn + leaky -> next bf16 activations; last layer also keeps f32 (reuse Agg)
        float* actf = (L == 2) ? Agg : nullptr;
        k_bn_act<<<dim3((unsigned)cdivll(nH, TB)), TB, 0, stream>>>(
            Hf, mu, rstd, ly.g, ly.be, actb, actf, nH, ly.dout);
    }

    // mean pool per graph + linear head + argmax
    k_pool_head<<<dim3(NGRAPH), 256, 0, stream>>>(Agg, Wp, bp, out);
}
